// model_76733885710690
// MI455X (gfx1250) — compile-verified, matrix operand order checked by reading, not run on hardware
//
#include <hip/hip_runtime.h>

// ---------------------------------------------------------------------------
// CDNA5 (gfx1250, wave32) implementation.
// Dominant cost: four [9216x1024]x[1024x1024] GEMMs -> v_wmma_f32_16x16x32_bf16.
// GEMM v2: 64x64 wave tile (16 WMMA / K-step), double-buffered LDS staged with
// global_load_async_to_lds_b128 + s_wait_asynccnt (CDNA5 async path).
// ---------------------------------------------------------------------------

typedef __attribute__((ext_vector_type(16))) __bf16 v16bf;
typedef __attribute__((ext_vector_type(8)))  float  v8f;

#define NR_   8
#define B_    256
#define EMB_  1024
#define OUT_  512
#define NEXT_ 36            // 8 + C(8,2)
#define MEXT_ (B_ * NEXT_)  // 9216 rows per side
#define MTOT_ (2 * MEXT_)   // 18432 (r_ext rows, then p_ext rows)

// triu_indices(8, k=1) order
__constant__ int PA[28] = {0,0,0,0,0,0,0,1,1,1,1,1,1,2,2,2,2,2,3,3,3,3,4,4,4,5,5,6};
__constant__ int PC[28] = {1,2,3,4,5,6,7,2,3,4,5,6,7,3,4,5,6,7,4,5,6,7,5,6,7,6,7,7};

struct Frag32 { uint4 lo, hi; };   // 32 bytes == v16bf

__device__ inline unsigned short f2bf(float x) {
    unsigned int u = __builtin_bit_cast(unsigned int, x);
    return (unsigned short)((u + 0x7FFFu + ((u >> 16) & 1u)) >> 16);  // RNE
}

// Per-lane async copy of 16 bytes global -> LDS (tracked by ASYNCcnt).
// Low 32 bits of a generic __shared__ pointer are the wave-relative LDS byte
// address (see ISA 10.2, apertures).
__device__ inline void async_b128(const unsigned short* g, unsigned short* l) {
    unsigned int loff = (unsigned int)(unsigned long long)l;
    asm volatile("global_load_async_to_lds_b128 %0, %1, off"
                 :: "v"(loff), "v"(g) : "memory");
}
__device__ inline void wait_async0() {
    asm volatile("s_wait_asynccnt 0x0" ::: "memory");
}

// ---------------------------------------------------------------------------
// Prep: W [K,N] f32  ->  Wt [N,K] bf16  (so B-fragments load contiguously)
// ---------------------------------------------------------------------------
__global__ __launch_bounds__(256) void transpose_bf16_kernel(
    const float* __restrict__ in, unsigned short* __restrict__ out, int K, int N) {
    size_t idx = (size_t)blockIdx.x * 256 + threadIdx.x;   // over N*K outputs
    int k = (int)(idx % K);
    int n = (int)(idx / K);
    out[idx] = f2bf(in[(size_t)k * N + n]);
}

// ---------------------------------------------------------------------------
// Prep: build bf16 extended features, rows = side*9216 + b*36 + i
// ---------------------------------------------------------------------------
__global__ __launch_bounds__(256) void build_ext_kernel(
    const float* __restrict__ r, const float* __restrict__ p,
    unsigned short* __restrict__ ext) {
    size_t idx = (size_t)blockIdx.x * 256 + threadIdx.x;   // < MTOT_*EMB_
    int d = (int)(idx & (EMB_ - 1));
    size_t row = idx >> 10;
    int side = row >= MEXT_;
    int rb = (int)(row - (size_t)side * MEXT_);
    int b = rb / NEXT_;
    int i = rb % NEXT_;
    const float* f = side ? p : r;                          // [NR, B, EMB]
    float v;
    if (i < 8) {
        v = f[((size_t)i * B_ + b) * EMB_ + d];
    } else {
        int a = PA[i - 8], c = PC[i - 8];
        v = f[((size_t)a * B_ + b) * EMB_ + d] + f[((size_t)c * B_ + b) * EMB_ + d];
    }
    ext[idx] = f2bf(v);
}

// ---------------------------------------------------------------------------
// Prep: diff = r - p, bf16, flat [8*256, 1024]
// ---------------------------------------------------------------------------
__global__ __launch_bounds__(256) void build_diff_kernel(
    const float* __restrict__ r, const float* __restrict__ p,
    unsigned short* __restrict__ dif) {
    size_t idx = (size_t)blockIdx.x * 256 + threadIdx.x;
    dif[idx] = f2bf(r[idx] - p[idx]);
}

// ---------------------------------------------------------------------------
// WMMA GEMM:  C[M,N] = alpha * (A[M,K] x Bt[N,K]^T) + bias[N]
// A, Bt bf16 row-major; C f32 row-major.
// Block: 256 threads = 8 waves (4 M x 2 N); block tile 256x128; wave tile
// 64x64 = 4x4 WMMA fragments. K stepped by 32. LDS double-buffered, filled by
// per-lane global_load_async_to_lds_b128; pitch 40 halfwords (bank spread).
// ---------------------------------------------------------------------------
__global__ __launch_bounds__(256) void gemm_bf16_kernel(
    const unsigned short* __restrict__ A,
    const unsigned short* __restrict__ Bt,
    float* __restrict__ C,
    const float* __restrict__ bias,
    int M, int N, int K, float alpha) {
    __shared__ alignas(16) unsigned short As[2][256 * 40];
    __shared__ alignas(16) unsigned short Bs[2][128 * 40];

    const int tid  = threadIdx.x;
    const int wave = tid >> 5;
    const int lane = tid & 31;
    const int half = lane >> 4;
    const int lr   = lane & 15;
    const int bm = blockIdx.x * 256;
    const int bn = blockIdx.y * 128;
    const int wm = (wave >> 1) * 64;   // 4 waves along M
    const int wn = (wave & 1) * 64;    // 2 waves along N

    // staging roles: A row = tid (32 elems = 4 x b128);
    //                B row = tid>>1, halfwords (tid&1)*16.. (2 x b128)
    const int brow = tid >> 1;
    const int bco  = (tid & 1) * 16;

    v8f acc[4][4] = {};

    const int nt = K / 32;
    // prologue: stage tile 0 into buffer 0
    {
        const unsigned short* gA = A + (size_t)(bm + tid) * K;
        #pragma unroll
        for (int c = 0; c < 4; ++c) async_b128(gA + c * 8, &As[0][tid * 40 + c * 8]);
        const unsigned short* gB = Bt + (size_t)(bn + brow) * K + bco;
        #pragma unroll
        for (int c = 0; c < 2; ++c) async_b128(gB + c * 8, &Bs[0][brow * 40 + bco + c * 8]);
    }

    for (int t = 0; t < nt; ++t) {
        const int cur = t & 1;
        wait_async0();      // my copies for tile t are done
        __syncthreads();    // everyone's copies done; prev iter reads done

        if (t + 1 < nt) {   // stage tile t+1 into the other buffer (overlaps compute)
            const int k0 = (t + 1) * 32;
            const unsigned short* gA = A + (size_t)(bm + tid) * K + k0;
            #pragma unroll
            for (int c = 0; c < 4; ++c)
                async_b128(gA + c * 8, &As[1 - cur][tid * 40 + c * 8]);
            const unsigned short* gB = Bt + (size_t)(bn + brow) * K + k0 + bco;
            #pragma unroll
            for (int c = 0; c < 2; ++c)
                async_b128(gB + c * 8, &Bs[1 - cur][brow * 40 + bco + c * 8]);
        }

        // A fragment 16x32: lanes 0-15 row=lr K{0..7,16..23}; lanes 16-31 K{8..15,24..31}
        v16bf afrag[4], bfrag[4];
        #pragma unroll
        for (int i = 0; i < 4; ++i) {
            int base = (wm + i * 16 + lr) * 40 + half * 8;
            Frag32 fr;
            fr.lo = *reinterpret_cast<const uint4*>(&As[cur][base]);
            fr.hi = *reinterpret_cast<const uint4*>(&As[cur][base + 16]);
            afrag[i] = __builtin_bit_cast(v16bf, fr);
        }
        #pragma unroll
        for (int j = 0; j < 4; ++j) {
            int base = (wn + j * 16 + lr) * 40 + half * 8;
            Frag32 fr;
            fr.lo = *reinterpret_cast<const uint4*>(&Bs[cur][base]);
            fr.hi = *reinterpret_cast<const uint4*>(&Bs[cur][base + 16]);
            bfrag[j] = __builtin_bit_cast(v16bf, fr);
        }

        #pragma unroll
        for (int i = 0; i < 4; ++i)
            #pragma unroll
            for (int j = 0; j < 4; ++j)
                acc[i][j] = __builtin_amdgcn_wmma_f32_16x16x32_bf16(
                    false, afrag[i], false, bfrag[j],
                    (short)0, acc[i][j], false, false);

        __syncthreads();    // fragment reads done before next overwrite of buf cur
    }

    // Epilogue: C/D layout: col = lane&15, VGPR r -> row r + (lane>=16)*8
    #pragma unroll
    for (int j = 0; j < 4; ++j) {
        int n = bn + wn + j * 16 + lr;
        float bv = bias ? bias[n] : 0.0f;
        #pragma unroll
        for (int i = 0; i < 4; ++i) {
            int mbase = bm + wm + i * 16 + half * 8;
            #pragma unroll
            for (int rr = 0; rr < 8; rr++)
                C[(size_t)(mbase + rr) * N + n] = acc[i][j][rr] * alpha + bv;
        }
    }
}

// ---------------------------------------------------------------------------
// Attention: per (dir, batch) block. scores[36][36] = Q.K (Q pre-scaled 1/32),
// softmax over k, mean over q, then collapse 36 att weights -> 8 effective
// per-original-row weights.  dir 0: q=p_ext, k=r_ext;  dir 1: q=r_ext, k=p_ext.
// ---------------------------------------------------------------------------
__global__ __launch_bounds__(256) void attention_kernel(
    const float* __restrict__ Q, const float* __restrict__ Kb,
    float* __restrict__ w_out) {
    __shared__ float s[NEXT_ * NEXT_];
    __shared__ float att[NEXT_];

    const int blk = blockIdx.x;
    const int b   = blk & (B_ - 1);
    const int dir = blk >> 8;
    const size_t qbase = ((dir == 0) ? (size_t)MEXT_ : 0) + (size_t)b * NEXT_;
    const size_t kbase = ((dir == 0) ? 0 : (size_t)MEXT_) + (size_t)b * NEXT_;

    const int tid = threadIdx.x, wave = tid >> 5, lane = tid & 31;

    for (int pr = wave; pr < NEXT_ * NEXT_; pr += 8) {
        int qi = pr / NEXT_, ki = pr % NEXT_;
        const float* qr = Q  + (qbase + qi) * EMB_;
        const float* kr = Kb + (kbase + ki) * EMB_;
        float sum = 0.0f;
        for (int d = lane; d < EMB_; d += 32) sum += qr[d] * kr[d];
        #pragma unroll
        for (int o = 16; o; o >>= 1) sum += __shfl_xor(sum, o, 32);
        if (lane == 0) s[pr] = sum;
    }
    __syncthreads();

    if (tid < NEXT_) {                       // softmax of row `tid`
        float mx = -3.0e38f;
        for (int k = 0; k < NEXT_; k++) mx = fmaxf(mx, s[tid * NEXT_ + k]);
        float sm = 0.0f;
        for (int k = 0; k < NEXT_; k++) {
            float e = __expf(s[tid * NEXT_ + k] - mx);
            s[tid * NEXT_ + k] = e; sm += e;
        }
        float inv = 1.0f / sm;
        for (int k = 0; k < NEXT_; k++) s[tid * NEXT_ + k] *= inv;
    }
    __syncthreads();

    if (tid < NEXT_) {                       // mean over query dim
        float a = 0.0f;
        for (int q = 0; q < NEXT_; q++) a += s[q * NEXT_ + tid];
        att[tid] = a * (1.0f / NEXT_);
    }
    __syncthreads();

    if (tid < NR_) {                         // collapse pair weights onto originals
        float w = att[tid];
        for (int pr = 0; pr < 28; pr++)
            if (PA[pr] == tid || PC[pr] == tid) w += att[8 + pr];
        w_out[(size_t)dir * B_ * NR_ + (size_t)b * NR_ + tid] = w;
    }
}

// ---------------------------------------------------------------------------
// reaction_vectors[b,d] = sum_i wr[b,i]*r[i,b,d] - wp[b,i]*p[i,b,d]
// ---------------------------------------------------------------------------
__global__ __launch_bounds__(256) void weighted_kernel(
    const float* __restrict__ r, const float* __restrict__ p,
    const float* __restrict__ w, float* __restrict__ out) {
    const int b = blockIdx.x;
    const int t = threadIdx.x;
    float wr[NR_], wp[NR_];
    #pragma unroll
    for (int i = 0; i < NR_; i++) {
        wr[i] = w[(size_t)b * NR_ + i];
        wp[i] = w[(size_t)B_ * NR_ + (size_t)b * NR_ + i];
    }
    #pragma unroll
    for (int j = 0; j < EMB_ / 256; j++) {
        int d = t + j * 256;
        float acc = 0.0f;
        #pragma unroll
        for (int i = 0; i < NR_; i++) {
            acc += wr[i] * r[((size_t)i * B_ + b) * EMB_ + d];
            acc -= wp[i] * p[((size_t)i * B_ + b) * EMB_ + d];
        }
        out[(size_t)b * EMB_ + d] = acc;
    }
}

// ---------------------------------------------------------------------------
extern "C" void kernel_launch(void* const* d_in, const int* in_sizes, int n_in,
                              void* d_out, int out_size, void* d_ws, size_t ws_size,
                              hipStream_t stream) {
    (void)in_sizes; (void)n_in; (void)out_size; (void)ws_size;

    const float* r  = (const float*)d_in[0];   // [8,256,1024]
    const float* p  = (const float*)d_in[1];   // [8,256,1024]
    const float* Wq = (const float*)d_in[2];   // [1024,1024]
    const float* Wk = (const float*)d_in[3];   // [1024,1024]
    const float* Wp = (const float*)d_in[4];   // [1024,512]
    const float* bp = (const float*)d_in[5];   // [512]

    // workspace layout (~190 MB)
    char* ws = (char*)d_ws;
    unsigned short* ext = (unsigned short*)ws; ws += (size_t)MTOT_ * EMB_ * 2;
    unsigned short* WqT = (unsigned short*)ws; ws += (size_t)EMB_ * EMB_ * 2;
    unsigned short* WkT = (unsigned short*)ws; ws += (size_t)EMB_ * EMB_ * 2;
    unsigned short* WpT = (unsigned short*)ws; ws += (size_t)OUT_ * EMB_ * 2;
    unsigned short* dif = (unsigned short*)ws; ws += (size_t)NR_ * B_ * EMB_ * 2;
    float* Qb = (float*)ws; ws += (size_t)MTOT_ * EMB_ * 4;
    float* Kb = (float*)ws; ws += (size_t)MTOT_ * EMB_ * 4;
    float* wv = (float*)ws; ws += (size_t)2 * B_ * NR_ * 4;

    float* out1 = (float*)d_out;                    // [8,256,512]
    float* out2 = out1 + (size_t)NR_ * B_ * OUT_;   // [256,1024]

    transpose_bf16_kernel<<<(EMB_ * EMB_) / 256, 256, 0, stream>>>(Wq, WqT, EMB_, EMB_);
    transpose_bf16_kernel<<<(EMB_ * EMB_) / 256, 256, 0, stream>>>(Wk, WkT, EMB_, EMB_);
    transpose_bf16_kernel<<<(EMB_ * OUT_) / 256, 256, 0, stream>>>(Wp, WpT, EMB_, OUT_);
    build_ext_kernel<<<((size_t)MTOT_ * EMB_) / 256, 256, 0, stream>>>(r, p, ext);
    build_diff_kernel<<<((size_t)NR_ * B_ * EMB_) / 256, 256, 0, stream>>>(r, p, dif);

    // Q = ext @ Wq, scores scale 1/sqrt(1024) folded in.  K = ext @ Wk.
    gemm_bf16_kernel<<<dim3(MTOT_ / 256, EMB_ / 128), 256, 0, stream>>>(
        ext, WqT, Qb, nullptr, MTOT_, EMB_, EMB_, 0.03125f);
    gemm_bf16_kernel<<<dim3(MTOT_ / 256, EMB_ / 128), 256, 0, stream>>>(
        ext, WkT, Kb, nullptr, MTOT_, EMB_, EMB_, 1.0f);
    // out = (r - p) @ W_pred + b_pred
    gemm_bf16_kernel<<<dim3((NR_ * B_) / 256, OUT_ / 128), 256, 0, stream>>>(
        dif, WpT, out1, bp, NR_ * B_, OUT_, EMB_, 1.0f);

    attention_kernel<<<2 * B_, 256, 0, stream>>>(Qb, Kb, wv);
    weighted_kernel<<<B_, 256, 0, stream>>>(r, p, wv, out2);
}
